// MultiHeadAttention_81758997447297
// MI455X (gfx1250) — matrix operand order repaired and checked by reading, not compiled, not run on hardware
//
#include <hip/hip_runtime.h>
#include <hip/hip_bf16.h>

// Multi-head attention forward for MI455X (gfx1250), wave32 + WMMA bf16.
// GEMMs: register-pipelined global->reg->bf16-LDS staging (convert once).
// Attention: async global->LDS V staging (ASYNCcnt) + bf16 prob tiles in LDS.
//
//   D_MODEL=1024, NUM_HEADS=16, D_K=64, B=2, S=2048, M_TOK=B*S=4096
//
// d_ws layout (floats): [q_heads 4.2M][k_heads 4.2M][v_heads 4.2M][concat 4.2M]
// d_out layout (floats): [out 4096*1024][attn 2*16*2048*2048]

#define D_MODEL 1024
#define NUM_HEADS 16
#define D_K 64
#define SEQ 2048
#define BATCH 2
#define M_TOK (BATCH * SEQ)

typedef __attribute__((ext_vector_type(16))) __bf16 v16bf;
typedef __attribute__((ext_vector_type(4))) __bf16 v4bf;
typedef __attribute__((ext_vector_type(8))) float v8f;

#if defined(__has_builtin)
#if __has_builtin(__builtin_amdgcn_global_load_async_to_lds_b128) && \
    __has_builtin(__builtin_amdgcn_s_wait_asynccnt)
#define HAVE_ASYNC_LDS 1
#endif
#endif

#ifdef HAVE_ASYNC_LDS
// Builtin signature (from hipcc diagnostic): (v4i AS1*, v4i AS3*, imm, imm)
typedef int v4i_gcc __attribute__((vector_size(16)));
typedef __attribute__((address_space(1))) v4i_gcc as1_v4i;
typedef __attribute__((address_space(3))) v4i_gcc as3_v4i;
#endif

// 16-byte global -> LDS copy: GLOBAL_LOAD_ASYNC_TO_LDS_B128 (ASYNCcnt) when
// available, else sync load + ds_store.
__device__ __forceinline__ void copy16_g2l(const float* g, float* l) {
#ifdef HAVE_ASYNC_LDS
  __builtin_amdgcn_global_load_async_to_lds_b128(
      (as1_v4i*)(unsigned long long)g,
      (as3_v4i*)(unsigned int)(unsigned long long)l, 0, 0);
#else
  *(float4*)l = *(const float4*)g;
#endif
}

__device__ __forceinline__ void wait_async_all() {
#ifdef HAVE_ASYNC_LDS
  __builtin_amdgcn_s_wait_asynccnt(0);
#endif
}

// 4 f32 -> 4 bf16 (v_cvt_pk_bf16_f32 x2) -> one ds_store_b64.
__device__ __forceinline__ void store4_bf(const float4 v, __bf16* l) {
  v4bf o;
  o[0] = (__bf16)v.x;
  o[1] = (__bf16)v.y;
  o[2] = (__bf16)v.z;
  o[3] = (__bf16)v.w;
  *(v4bf*)l = o;
}

// ---------------------------------------------------------------------------
// WMMA wrapper: D = A(16x32 bf16) * B(32x16 bf16) + C(16x16 f32)
// ---------------------------------------------------------------------------
__device__ __forceinline__ v8f wmma_bf16(v16bf a, v16bf b, v8f c) {
  return __builtin_amdgcn_wmma_f32_16x16x32_bf16(
      false, a, false, b, (short)0, c, false, false);
}

// ---------------------------------------------------------------------------
// A-fragment (16x32, MxK) from a row-major bf16 source (LDS), contiguous k order:
// lanes 0-15: row M=lane, K=0..15; lanes 16-31: K=16..31.
// Two 8-element contiguous bf16 runs -> 2x ds_load_b128, no converts.
// ---------------------------------------------------------------------------
__device__ __forceinline__ v16bf load_a_bf(const __bf16* src, int lda) {
  const int lane = threadIdx.x & 31;
  const int half = lane >> 4;
  const int row = lane & 15;
  const __bf16* p = src + row * lda;
  const int k1 = half * 16;
  const int k2 = half * 16 + 8;
  v16bf a;
#pragma unroll
  for (int i = 0; i < 8; ++i) {
    a[i] = p[k1 + i];
    a[8 + i] = p[k2 + i];
  }
  return a;
}

// A-fragment from a row-major f32 source (global), with convert.
__device__ __forceinline__ v16bf load_a_f32(const float* src, int lda) {
  const int lane = threadIdx.x & 31;
  const int half = lane >> 4;
  const int row = lane & 15;
  const float* p = src + row * lda;
  const int k1 = half * 16;
  const int k2 = half * 16 + 8;
  v16bf a;
#pragma unroll
  for (int i = 0; i < 8; ++i) {
    a[i] = (__bf16)p[k1 + i];
    a[8 + i] = (__bf16)p[k2 + i];
  }
  return a;
}

// ---------------------------------------------------------------------------
// B-fragment (32x16, KxN) with B[k][n] = W[n][k], W row-major [N,K] bf16
// (LDS tile): 16 contiguous bf16 per lane -> 2x ds_load_b128, no converts.
// ---------------------------------------------------------------------------
__device__ __forceinline__ v16bf load_b_bf(const __bf16* W, int ldw, int n0) {
  const int lane = threadIdx.x & 31;
  const int half = lane >> 4;
  const int n = n0 + (lane & 15);
  const __bf16* p = W + n * ldw + half * 16;
  v16bf b;
#pragma unroll
  for (int i = 0; i < 16; ++i) b[i] = p[i];
  return b;
}

// B-fragment with B[k][n] = W[n][k] from a row-major [N,K] f32 source.
__device__ __forceinline__ v16bf load_b_nk_f32(const float* W, int ldw, int n0,
                                               int k0) {
  const int lane = threadIdx.x & 31;
  const int half = lane >> 4;
  const int n = n0 + (lane & 15);
  const float* p = W + (size_t)n * ldw + k0 + half * 16;
  v16bf b;
#pragma unroll
  for (int i = 0; i < 16; ++i) b[i] = (__bf16)p[i];
  return b;
}

// B-fragment (32x16, KxN) from a row-major [K,N] f32 source (V tile in LDS).
__device__ __forceinline__ v16bf load_b_kn_f32(const float* src, int ldn,
                                               int n0) {
  const int lane = threadIdx.x & 31;
  const int half = lane >> 4;
  const int n = n0 + (lane & 15);
  const float* p = src + (half * 16) * ldn + n;
  v16bf b;
#pragma unroll
  for (int i = 0; i < 16; ++i) b[i] = (__bf16)p[i * ldn];
  return b;
}

// ---------------------------------------------------------------------------
// GEMM + bias:  Y[m,n] = sum_k X[m,k] * W[n,k] + bias[n]
//   mode 0: Y row-major [M,N];  mode 1: head-split (b,h,s,d)
// 128(M) x 64(N) tile per 256-thread block. bf16 tiles in LDS, converted once
// at staging; next k-step's global loads are issued before the WMMAs and
// converted/stored after them (latency hidden behind compute).
// ---------------------------------------------------------------------------
__global__ void __launch_bounds__(256, 1) gemm_bias_kernel(
    const float* __restrict__ X, const float* __restrict__ W,
    const float* __restrict__ bias, float* __restrict__ Y, int M, int N, int K,
    int mode) {
  __shared__ __bf16 xs[2][128 * 32];  // 16 KB
  __shared__ __bf16 wt[2][64 * 32];   // 8 KB

  const int wave = threadIdx.x >> 5;
  const int lane = threadIdx.x & 31;
  const int half = lane >> 4;
  const int mrow0 = blockIdx.y * 128;
  const int n0 = blockIdx.x * 64;
  const int nk = K >> 5;

  v8f acc[4] = {};

  // Prologue: stage kb = 0 (load + convert + ds_store).
  for (int c = threadIdx.x; c < 1024; c += 256) {
    const int row = c >> 3, col = (c & 7) * 4;
    store4_bf(*(const float4*)(X + (size_t)(mrow0 + row) * K + col),
              &xs[0][row * 32 + col]);
  }
  for (int c = threadIdx.x; c < 512; c += 256) {
    const int row = c >> 3, col = (c & 7) * 4;
    store4_bf(*(const float4*)(W + (size_t)(n0 + row) * K + col),
              &wt[0][row * 32 + col]);
  }
  __syncthreads();

  for (int kb = 0; kb < nk; ++kb) {
    const int cur = kb & 1;
    const bool more = (kb + 1 < nk);
    float4 rx[4], rw[2];
    if (more) {  // issue next k-step's global loads (results used after wmma)
      const int kcol = (kb + 1) * 32;
#pragma unroll
      for (int i = 0; i < 4; ++i) {
        const int c = threadIdx.x + i * 256;
        const int row = c >> 3, col = (c & 7) * 4;
        rx[i] = *(const float4*)(X + (size_t)(mrow0 + row) * K + kcol + col);
      }
#pragma unroll
      for (int i = 0; i < 2; ++i) {
        const int c = threadIdx.x + i * 256;
        const int row = c >> 3, col = (c & 7) * 4;
        rw[i] = *(const float4*)(W + (size_t)(n0 + row) * K + kcol + col);
      }
    }
    // Compute kb: 10 ds_load_b128 + 4 wmma per wave, no converts.
    v16bf a = load_a_bf(&xs[cur][wave * 16 * 32], 32);
#pragma unroll
    for (int t = 0; t < 4; ++t) {
      v16bf b = load_b_bf(&wt[cur][0], 32, t * 16);
      acc[t] = wmma_bf16(a, b, acc[t]);
    }
    if (more) {  // convert + store the staged registers into the other buffer
#pragma unroll
      for (int i = 0; i < 4; ++i) {
        const int c = threadIdx.x + i * 256;
        const int row = c >> 3, col = (c & 7) * 4;
        store4_bf(rx[i], &xs[cur ^ 1][row * 32 + col]);
      }
#pragma unroll
      for (int i = 0; i < 2; ++i) {
        const int c = threadIdx.x + i * 256;
        const int row = c >> 3, col = (c & 7) * 4;
        store4_bf(rw[i], &wt[cur ^ 1][row * 32 + col]);
      }
    }
    __syncthreads();
  }

  const int ncol = lane & 15;
#pragma unroll
  for (int t = 0; t < 4; ++t) {
    const int n = n0 + t * 16 + ncol;
    const float bv = bias[n];
#pragma unroll
    for (int r = 0; r < 8; ++r) {
      const int m = mrow0 + wave * 16 + half * 8 + r;
      const float val = acc[t][r] + bv;
      if (mode == 0) {
        Y[(size_t)m * N + n] = val;
      } else {
        const int bb = m >> 11, s = m & (SEQ - 1);
        const int hh = n >> 6, d = n & (D_K - 1);
        Y[(((size_t)bb * NUM_HEADS + hh) * SEQ + s) * D_K + d] = val;
      }
    }
  }
}

// ---------------------------------------------------------------------------
// Fused attention: one block per (16-row block, head, batch).
// Phase 1: scores(16x2048) = q_blk @ k^T / 8 -> LDS f32 (128 KB)
// Phase 2: row softmax (float4); writes f32 attn to global (b128) AND a bf16
//          copy of the probs into LDS (64 KB) for phase 3's A-fragments.
// Phase 3: out(16x64) = probs @ v; V tiles async-double-buffered in LDS.
// Dynamic LDS: 128 KB + 64 KB + 16 KB = 208 KB (<= 320 KB/WGP on CDNA5).
// ---------------------------------------------------------------------------
__global__ void __launch_bounds__(256, 1) attention_kernel(
    const float* __restrict__ qh, const float* __restrict__ kh,
    const float* __restrict__ vh, float* __restrict__ attn,
    float* __restrict__ concat) {
  extern __shared__ char smemraw[];
  float* sc = (float*)smemraw;                             // 16*SEQ f32
  __bf16* sbf = (__bf16*)(smemraw + 16 * SEQ * 4);         // 16*SEQ bf16
  float* vbuf0 = (float*)(smemraw + 16 * SEQ * 6);         // 32*64 f32
  float* vbuf1 = vbuf0 + 32 * D_K;                         // 32*64 f32

  const int b = blockIdx.z;
  const int h = blockIdx.y;
  const int rb = blockIdx.x;
  const int wave = threadIdx.x >> 5;
  const int lane = threadIdx.x & 31;
  const int half = lane >> 4;

  const size_t headoff = ((size_t)b * NUM_HEADS + h) * SEQ * D_K;
  const float* q0 = qh + headoff + (size_t)rb * 16 * D_K;  // [16 x 64]
  const float* kp = kh + headoff;                          // [2048 x 64]
  const float* vp = vh + headoff;                          // [2048 x 64]

  // Kick off the first V tile copy; overlaps phases 1-2 on the async path.
  for (int c = threadIdx.x; c < 512; c += 256) {
    const int row = c >> 4;  // 16 chunks per 64-float V row
    const int col = (c & 15) * 4;
    copy16_g2l(vp + (size_t)row * D_K + col, vbuf0 + row * D_K + col);
  }

  // ---- Phase 1: scores. Wave w covers key columns [w*256, w*256+256). ----
  const v16bf aq0 = load_a_f32(q0, D_K);       // d = 0..31
  const v16bf aq1 = load_a_f32(q0 + 32, D_K);  // d = 32..63
  for (int nt = 0; nt < 16; ++nt) {
    const int n0 = wave * 256 + nt * 16;
    if (nt + 1 < 16)  // global_prefetch_b8 of the next K-row block
      __builtin_prefetch(kp + (size_t)(n0 + 16) * D_K, 0, 1);
    v8f acc = {};
    v16bf bk0 = load_b_nk_f32(kp, D_K, n0, 0);
    acc = wmma_bf16(aq0, bk0, acc);
    v16bf bk1 = load_b_nk_f32(kp, D_K, n0, 32);
    acc = wmma_bf16(aq1, bk1, acc);
    const int ncol = n0 + (lane & 15);
#pragma unroll
    for (int r = 0; r < 8; ++r)
      sc[(half * 8 + r) * SEQ + ncol] = acc[r] * 0.125f;  // 1/sqrt(64)
  }
  __syncthreads();

  // ---- Phase 2: softmax, one wave per 2 rows, float4 all the way. ----
  for (int rr = 0; rr < 2; ++rr) {
    const int row = wave * 2 + rr;
    float4* srow = (float4*)(sc + row * SEQ);
    float mx = -3.402823466e38f;
    for (int c = lane; c < SEQ / 4; c += 32) {
      const float4 x = srow[c];
      mx = fmaxf(fmaxf(fmaxf(mx, x.x), fmaxf(x.y, x.z)), x.w);
    }
#pragma unroll
    for (int off = 16; off > 0; off >>= 1)
      mx = fmaxf(mx, __shfl_xor(mx, off, 32));
    float sum = 0.f;
    for (int c = lane; c < SEQ / 4; c += 32) {
      float4 x = srow[c];
      x.x = __expf(x.x - mx);
      x.y = __expf(x.y - mx);
      x.z = __expf(x.z - mx);
      x.w = __expf(x.w - mx);
      srow[c] = x;
      sum += (x.x + x.y) + (x.z + x.w);
    }
#pragma unroll
    for (int off = 16; off > 0; off >>= 1) sum += __shfl_xor(sum, off, 32);
    const float inv = 1.f / sum;
    float4* arow = (float4*)(attn + (((size_t)b * NUM_HEADS + h) * SEQ +
                                     rb * 16 + row) *
                                        SEQ);
    __bf16* brow = sbf + row * SEQ;
    for (int c = lane; c < SEQ / 4; c += 32) {
      float4 x = srow[c];
      x.x *= inv;
      x.y *= inv;
      x.z *= inv;
      x.w *= inv;
      arow[c] = x;                 // b128 store of the f32 attn tensor
      store4_bf(x, brow + 4 * c);  // bf16 probs for phase 3 (no re-convert)
    }
  }

  // ---- Phase 3: out = probs @ v with async-double-buffered V tiles. ----
  v8f acc = {};
  for (int ks = 0; ks < SEQ / 32; ++ks) {
    const int cur = ks & 1;
    float* vcur = cur ? vbuf1 : vbuf0;
    float* vnxt = cur ? vbuf0 : vbuf1;
    wait_async_all();  // V tile for vcur landed
    __syncthreads();   // visible to all waves; vnxt free to overwrite
    if (ks + 1 < SEQ / 32) {
      for (int c = threadIdx.x; c < 512; c += 256) {
        const int row = c >> 4;
        const int col = (c & 15) * 4;
        copy16_g2l(vp + (size_t)((ks + 1) * 32 + row) * D_K + col,
                   vnxt + row * D_K + col);
      }
    }
    if (wave < 4) {
      v16bf a = load_a_bf(sbf + ks * 32, SEQ);  // 2x ds_load_b128, no cvt
      v16bf bv = load_b_kn_f32(vcur, D_K, wave * 16);
      acc = wmma_bf16(a, bv, acc);
    }
  }
  if (wave < 4) {
    const int ncol = lane & 15;
#pragma unroll
    for (int r = 0; r < 8; ++r) {
      const int m = b * SEQ + rb * 16 + half * 8 + r;
      const int col = h * D_K + wave * 16 + ncol;
      concat[(size_t)m * D_MODEL + col] = acc[r];
    }
  }
}

// ---------------------------------------------------------------------------
extern "C" void kernel_launch(void* const* d_in, const int* in_sizes, int n_in,
                              void* d_out, int out_size, void* d_ws,
                              size_t ws_size, hipStream_t stream) {
  const float* Q = (const float*)d_in[0];
  const float* Kin = (const float*)d_in[1];
  const float* Vin = (const float*)d_in[2];
  const float* Wq = (const float*)d_in[3];
  const float* Wk = (const float*)d_in[4];
  const float* Wv = (const float*)d_in[5];
  const float* Wo = (const float*)d_in[6];
  const float* bq = (const float*)d_in[7];
  const float* bk = (const float*)d_in[8];
  const float* bv = (const float*)d_in[9];
  const float* bo = (const float*)d_in[10];

  float* out = (float*)d_out;
  float* attn = out + (size_t)M_TOK * D_MODEL;

  const size_t headN = (size_t)M_TOK * D_MODEL;  // 4,194,304 floats
  float* ws = (float*)d_ws;
  float* q_heads = ws;
  float* k_heads = ws + headN;
  float* v_heads = ws + 2 * headN;
  float* concat = ws + 3 * headN;

  const dim3 blk(256, 1, 1);
  const dim3 gproj(D_MODEL / 64, M_TOK / 128, 1);  // (16, 32)

  gemm_bias_kernel<<<gproj, blk, 0, stream>>>(Q, Wq, bq, q_heads, M_TOK,
                                              D_MODEL, D_MODEL, 1);
  gemm_bias_kernel<<<gproj, blk, 0, stream>>>(Kin, Wk, bk, k_heads, M_TOK,
                                              D_MODEL, D_MODEL, 1);
  gemm_bias_kernel<<<gproj, blk, 0, stream>>>(Vin, Wv, bv, v_heads, M_TOK,
                                              D_MODEL, D_MODEL, 1);

  const dim3 gattn(SEQ / 16, NUM_HEADS, BATCH);  // (128, 16, 2)
  const size_t attn_lds = 16 * SEQ * 6 + 2 * 32 * D_K * 4;  // 208 KB
  attention_kernel<<<gattn, blk, attn_lds, stream>>>(q_heads, k_heads, v_heads,
                                                     attn, concat);

  gemm_bias_kernel<<<gproj, blk, 0, stream>>>(concat, Wo, bo, out, M_TOK,
                                              D_MODEL, D_MODEL, 0);
}
